// FastSelfAttention_15874199126557
// MI455X (gfx1250) — matrix operand order repaired and checked by reading, not compiled, not run on hardware
//
#include <hip/hip_runtime.h>
#include <hip/hip_bf16.h>

typedef __bf16 bf16_t;
typedef __attribute__((ext_vector_type(16))) __bf16 v16bf;
typedef __attribute__((ext_vector_type(8)))  float  v8f;

struct __align__(16) U128 { unsigned int x, y, z, w; };
union AFrag { v16bf v; U128 q[2]; };

#define BM 128
#define BN 128
#define BK 32
#define TPB 256

// ---------------------------------------------------------------------------
// Main bf16 WMMA GEMM:  out = (A[MxK] @ WT^T + bias) * alpha + rowadd + residual
//   A:  bf16 row-major M x K.
//   WT: bf16 row-major Npad x K (pre-transposed weight, zero-padded to >= BN
//       rows per N-tile so the hot loop needs no bounds check).
//   Tiles stream global -> LDS via CDNA5 async copies (ASYNCcnt), double
//   buffered; compute is v_wmma_f32_16x16x32_bf16 with fp32 accumulation.
// ---------------------------------------------------------------------------
__global__ __launch_bounds__(TPB) void gemm_bf16_wmma(
    const bf16_t* __restrict__ A, const bf16_t* __restrict__ WT,
    const float* __restrict__ bias, int M, int K, int Nw, float alpha,
    const float* __restrict__ rowadd, const float* __restrict__ residual,
    float* __restrict__ outF, bf16_t* __restrict__ outB,
    int scoreLayout, int Nseq)
{
    (void)M;
    __shared__ __align__(16) bf16_t As[2][BM][BK];
    __shared__ __align__(16) bf16_t Bs[2][BN][BK];

    const int tid   = threadIdx.x;
    const int lane  = tid & 31;
    const int wave  = tid >> 5;
    const int waveM = wave & 3;   // 4 wave-rows of 32 M
    const int waveN = wave >> 2;  // 2 wave-cols of 64 N
    const int mBase = blockIdx.x * BM;
    const int nBase = blockIdx.y * BN;

    v8f zero = {0.f, 0.f, 0.f, 0.f, 0.f, 0.f, 0.f, 0.f};
    v8f acc[2][4];
#pragma unroll
    for (int mi = 0; mi < 2; ++mi)
#pragma unroll
        for (int ni = 0; ni < 4; ++ni) acc[mi][ni] = zero;

    const int nk = K / BK;

    // Per-thread 16B chunk coordinates: 512 chunks per tile, 2 per thread.
    const int i0 = tid,        ar0 = i0 >> 2, ac0 = i0 & 3;
    const int i1 = tid + TPB,  ar1 = i1 >> 2, ac1 = i1 & 3;

    const unsigned long long agp0 = (unsigned long long)(A  + (size_t)(mBase + ar0) * K + ac0 * 8);
    const unsigned long long agp1 = (unsigned long long)(A  + (size_t)(mBase + ar1) * K + ac1 * 8);
    const unsigned long long bgp0 = (unsigned long long)(WT + (size_t)(nBase + ar0) * K + ac0 * 8);
    const unsigned long long bgp1 = (unsigned long long)(WT + (size_t)(nBase + ar1) * K + ac1 * 8);

    // 32-bit LDS byte offsets (low 32 bits of the flat address of LDS objects)
    const unsigned aLds0 = (unsigned)(unsigned long long)&As[0][ar0][ac0 * 8];
    const unsigned aLds1 = (unsigned)(unsigned long long)&As[0][ar1][ac1 * 8];
    const unsigned bLds0 = (unsigned)(unsigned long long)&Bs[0][ar0][ac0 * 8];
    const unsigned bLds1 = (unsigned)(unsigned long long)&Bs[0][ar1][ac1 * 8];
    const unsigned bufStride = BM * BK * 2;  // 8192 bytes per buffer

    auto issueTile = [&](int buf, int kt) {
        const unsigned long long koff = (unsigned long long)kt * (BK * 2);  // bytes
        const unsigned bo = (unsigned)buf * bufStride;
        asm volatile("global_load_async_to_lds_b128 %0, %1, off"
                     :: "v"(aLds0 + bo), "v"(agp0 + koff) : "memory");
        asm volatile("global_load_async_to_lds_b128 %0, %1, off"
                     :: "v"(aLds1 + bo), "v"(agp1 + koff) : "memory");
        asm volatile("global_load_async_to_lds_b128 %0, %1, off"
                     :: "v"(bLds0 + bo), "v"(bgp0 + koff) : "memory");
        asm volatile("global_load_async_to_lds_b128 %0, %1, off"
                     :: "v"(bLds1 + bo), "v"(bgp1 + koff) : "memory");
    };

    issueTile(0, 0);
    asm volatile("s_wait_asynccnt 0x0" ::: "memory");
    __syncthreads();

    const int half = (lane & 16) ? 1 : 0;

    for (int kt = 0; kt < nk; ++kt) {
        const int cur = kt & 1;
        if (kt + 1 < nk) issueTile(cur ^ 1, kt + 1);   // overlap with compute

        AFrag a[2], b[4];
#pragma unroll
        for (int mi = 0; mi < 2; ++mi) {
            // ISA 16-bit A 16x32 layout: lane half picks K {0..7,16..23} / {8..15,24..31}
            int r = waveM * 32 + mi * 16 + (lane & 15);
            const bf16_t* p = &As[cur][r][half * 8];
            a[mi].q[0] = *(const U128*)p;
            a[mi].q[1] = *(const U128*)(p + 16);
        }
#pragma unroll
        for (int ni = 0; ni < 4; ++ni) {
            // B 32x16: lane half picks K {0..7,16..23} / {8..15,24..31} (same as A)
            int r = waveN * 64 + ni * 16 + (lane & 15);
            const bf16_t* p = &Bs[cur][r][half * 8];
            b[ni].q[0] = *(const U128*)p;
            b[ni].q[1] = *(const U128*)(p + 16);
        }

#pragma unroll
        for (int mi = 0; mi < 2; ++mi)
#pragma unroll
            for (int ni = 0; ni < 4; ++ni)
                acc[mi][ni] = __builtin_amdgcn_wmma_f32_16x16x32_bf16(
                    false, a[mi].v, false, b[ni].v, (short)0, acc[mi][ni],
                    false, false);

        // Own async transfers done, then workgroup-wide visibility.
        asm volatile("s_wait_asynccnt 0x0" ::: "memory");
        __syncthreads();
    }

    // Epilogue. C/D layout: lane<16 -> N=lane, M=r; lane>=16 -> N=lane-16, M=8+r
    const int n_l   = lane & 15;
    const int m_off = (lane & 16) ? 8 : 0;
#pragma unroll
    for (int mi = 0; mi < 2; ++mi) {
#pragma unroll
        for (int ni = 0; ni < 4; ++ni) {
            const int n = nBase + waveN * 64 + ni * 16 + n_l;
            if (n >= Nw) continue;
#pragma unroll
            for (int r = 0; r < 8; ++r) {
                const int m = mBase + waveM * 32 + mi * 16 + m_off + r;
                float v = acc[mi][ni][r];
                if (bias)     v += bias[n];
                v *= alpha;
                if (rowadd)   v += rowadd[m];                       // mask[b*Nseq+nseq] == mask[m]
                if (residual) v += residual[(size_t)m * Nw + n];
                if (outF) {
                    size_t oidx;
                    if (scoreLayout) {
                        int bb = m / Nseq, ns = m % Nseq;           // (b, h*g, n) layout
                        oidx = ((size_t)(bb * Nw + n)) * Nseq + ns;
                    } else {
                        oidx = (size_t)m * Nw + n;
                    }
                    outF[oidx] = v;
                }
                if (outB) outB[(size_t)m * Nw + n] = (bf16_t)v;
            }
        }
    }
}

// --------------------------- prep kernels ----------------------------------
__global__ __launch_bounds__(TPB) void conv_f32_bf16(
    const float* __restrict__ s, bf16_t* __restrict__ d, size_t n)
{
    size_t i = (size_t)blockIdx.x * TPB + threadIdx.x;
    if (i < n) d[i] = (bf16_t)s[i];
}

// src: fp32 K x Nw (row-major) -> dst: bf16 Npad x K (transposed, zero-padded)
__global__ __launch_bounds__(TPB) void conv_transpose_w(
    const float* __restrict__ src, bf16_t* __restrict__ dst, int K, int Nw, int Npad)
{
    size_t idx = (size_t)blockIdx.x * TPB + threadIdx.x;
    if (idx >= (size_t)K * Npad) return;
    int k = (int)(idx / Npad), n = (int)(idx % Npad);
    float v = (n < Nw) ? src[(size_t)k * Nw + n] : 0.f;
    dst[(size_t)n * K + k] = (bf16_t)v;
}

// --------------------------- softmax over n --------------------------------
__global__ __launch_bounds__(TPB) void softmax_rows(float* __restrict__ x, int ncols)
{
    __shared__ float sred[8];
    const int tid = threadIdx.x;
    float* p = x + (size_t)blockIdx.x * ncols;

    float mx = -3.4e38f;
    for (int i = tid; i < ncols; i += TPB) mx = fmaxf(mx, p[i]);
#pragma unroll
    for (int o = 16; o > 0; o >>= 1) mx = fmaxf(mx, __shfl_xor(mx, o, 32));
    if ((tid & 31) == 0) sred[tid >> 5] = mx;
    __syncthreads();
    float bmax = sred[0];
#pragma unroll
    for (int w = 1; w < 8; ++w) bmax = fmaxf(bmax, sred[w]);
    __syncthreads();

    float s = 0.f;
    for (int i = tid; i < ncols; i += TPB) {
        float e = __expf(p[i] - bmax);
        p[i] = e;
        s += e;
    }
#pragma unroll
    for (int o = 16; o > 0; o >>= 1) s += __shfl_xor(s, o, 32);
    if ((tid & 31) == 0) sred[tid >> 5] = s;
    __syncthreads();
    float bs = 0.f;
#pragma unroll
    for (int w = 0; w < 8; ++w) bs += sred[w];
    float inv = 1.0f / bs;
    for (int i = tid; i < ncols; i += TPB) p[i] *= inv;
}

// ---------------- einsum 'bhgn,bhnd->bhdg' (G=4, D=64, H=16, DM=1024) ------
__global__ __launch_bounds__(TPB) void einsum_gdg(
    const float* __restrict__ w,    // (b*H + h, g, n)    contiguous n
    const float* __restrict__ src,  // (b, n, h*64 + d)   fp32
    float* __restrict__ out,        // (b*H + h, d, g)
    int Nseq)
{
    const int bh = blockIdx.x;      // 0 .. B*H-1
    const int b  = bh >> 4;
    const int h  = bh & 15;
    const int g  = threadIdx.x >> 6;  // 0..3
    const int d  = threadIdx.x & 63;  // 0..63
    const float* wp = w + ((size_t)bh * 4 + g) * Nseq;
    const float* sp = src + (size_t)b * Nseq * 1024 + h * 64 + d;
    float acc = 0.f;
#pragma unroll 8
    for (int n = 0; n < Nseq; ++n) acc += wp[n] * sp[(size_t)n * 1024];
    out[((size_t)bh * 64 + d) * 4 + g] = acc;
}

// ------------- out[b,n,c] = bf16( max_g( scal[b,n,c] * gmat[b,c,g] ) ) -----
__global__ __launch_bounds__(TPB) void maxcombine(
    const float* __restrict__ scal,   // (b, n, 1024)
    const float* __restrict__ gmat,   // (b, 1024, 4)
    bf16_t* __restrict__ out,         // (b, n, 1024) bf16
    int Nseq)
{
    size_t idx = (size_t)blockIdx.x * TPB + threadIdx.x;  // over B*Nseq*1024
    int col = (int)(idx & 1023);
    size_t row = idx >> 10;
    int b = (int)(row / Nseq);
    const float* gp = gmat + ((size_t)b * 1024 + col) * 4;
    float s = scal[idx];
    float m = s * gp[0];
    m = fmaxf(m, s * gp[1]);
    m = fmaxf(m, s * gp[2]);
    m = fmaxf(m, s * gp[3]);
    out[idx] = (bf16_t)m;
}

// ---------------------------------------------------------------------------
extern "C" void kernel_launch(void* const* d_in, const int* in_sizes, int n_in,
                              void* d_out, int out_size, void* d_ws, size_t ws_size,
                              hipStream_t stream)
{
    (void)in_sizes; (void)n_in; (void)out_size; (void)ws_size;
    const float* X    = (const float*)d_in[0];
    const float* mask = (const float*)d_in[1];
    const float* Wq   = (const float*)d_in[2];
    const float* bq   = (const float*)d_in[3];
    const float* Wqa  = (const float*)d_in[4];
    const float* bqa  = (const float*)d_in[5];
    const float* Wk   = (const float*)d_in[6];
    const float* bk   = (const float*)d_in[7];
    const float* Wka  = (const float*)d_in[8];
    const float* bka  = (const float*)d_in[9];
    const float* Wv   = (const float*)d_in[10];
    const float* bv   = (const float*)d_in[11];
    const float* Wt   = (const float*)d_in[12];
    const float* bt   = (const float*)d_in[13];

    const int Bb = 8, Nseq = 4096, DM = 1024;
    const int M = Bb * Nseq;        // 32768
    const int K = DM;               // 1024
    const float inv_scale = 0.125f; // 1/sqrt(64)

    char* ws = (char*)d_ws;
    size_t off = 0;
    auto carve = [&](size_t bytes) -> void* {
        void* p = ws + off;
        off += (bytes + 255) & ~(size_t)255;
        return p;
    };
    bf16_t* Xb   = (bf16_t*)carve((size_t)M * DM * 2);
    bf16_t* WqT  = (bf16_t*)carve((size_t)DM * DM * 2);
    bf16_t* WkT  = (bf16_t*)carve((size_t)DM * DM * 2);
    bf16_t* WvT  = (bf16_t*)carve((size_t)DM * DM * 2);
    bf16_t* WtT  = (bf16_t*)carve((size_t)DM * DM * 2);
    bf16_t* WqaT = (bf16_t*)carve((size_t)BN * DM * 2);   // 64 rows + zero pad to 128
    bf16_t* WkaT = (bf16_t*)carve((size_t)BN * DM * 2);
    float*  qf   = (float*)carve((size_t)M * DM * 4);
    bf16_t* qb   = (bf16_t*)carve((size_t)M * DM * 2);
    float*  kf   = (float*)carve((size_t)M * DM * 4);
    float*  vf   = (float*)carve((size_t)M * DM * 4);
    float*  qs   = (float*)carve((size_t)Bb * 64 * Nseq * 4);
    float*  gq   = (float*)carve((size_t)Bb * 1024 * 4 * 4);
    bf16_t* qkb  = (bf16_t*)carve((size_t)M * DM * 2);
    float*  ksc  = (float*)carve((size_t)Bb * 64 * Nseq * 4);
    float*  gk   = (float*)carve((size_t)Bb * 1024 * 4 * 4);
    bf16_t* ub   = (bf16_t*)carve((size_t)M * DM * 2);

    // ---- prep: bf16 conversions / weight transposes (zero-padded for Nw=64) ----
    conv_f32_bf16<<<(unsigned)(((size_t)M * DM + TPB - 1) / TPB), TPB, 0, stream>>>(X, Xb, (size_t)M * DM);
    conv_transpose_w<<<(DM * DM  + TPB - 1) / TPB, TPB, 0, stream>>>(Wq,  WqT,  DM, DM, DM);
    conv_transpose_w<<<(DM * DM  + TPB - 1) / TPB, TPB, 0, stream>>>(Wk,  WkT,  DM, DM, DM);
    conv_transpose_w<<<(DM * DM  + TPB - 1) / TPB, TPB, 0, stream>>>(Wv,  WvT,  DM, DM, DM);
    conv_transpose_w<<<(DM * DM  + TPB - 1) / TPB, TPB, 0, stream>>>(Wt,  WtT,  DM, DM, DM);
    conv_transpose_w<<<(DM * BN  + TPB - 1) / TPB, TPB, 0, stream>>>(Wqa, WqaT, DM, 64, BN);
    conv_transpose_w<<<(DM * BN  + TPB - 1) / TPB, TPB, 0, stream>>>(Wka, WkaT, DM, 64, BN);

    dim3 gridBig(M / BM, DM / BN);     // 256 x 8
    dim3 gridSm(M / BM, 1);            // Nw = 64 (padded weight rows)

    // ---- q, k, v projections ----
    gemm_bf16_wmma<<<gridBig, TPB, 0, stream>>>(Xb, WqT, bq, M, K, DM, 1.0f,
        nullptr, nullptr, qf, qb, 0, Nseq);
    gemm_bf16_wmma<<<gridBig, TPB, 0, stream>>>(Xb, WkT, bk, M, K, DM, 1.0f,
        nullptr, nullptr, kf, nullptr, 0, Nseq);
    gemm_bf16_wmma<<<gridBig, TPB, 0, stream>>>(Xb, WvT, bv, M, K, DM, 1.0f,
        nullptr, nullptr, vf, nullptr, 0, Nseq);

    // ---- qs = (q @ Wqa + bqa)/scale + mask  -> (b, h*g, n), then softmax(n) ----
    gemm_bf16_wmma<<<gridSm, TPB, 0, stream>>>(qb, WqaT, bqa, M, K, 64, inv_scale,
        mask, nullptr, qs, nullptr, 1, Nseq);
    softmax_rows<<<Bb * 64, TPB, 0, stream>>>(qs, Nseq);

    // ---- gq = einsum(qw, q) ; qk = max_g(k * gq) (bf16) ----
    einsum_gdg<<<Bb * 16, TPB, 0, stream>>>(qs, qf, gq, Nseq);
    maxcombine<<<(unsigned)(((size_t)M * DM) / TPB), TPB, 0, stream>>>(kf, gq, qkb, Nseq);

    // ---- ksc = (qk @ Wka + bka)/scale + mask (NO softmax) ----
    gemm_bf16_wmma<<<gridSm, TPB, 0, stream>>>(qkb, WkaT, bka, M, K, 64, inv_scale,
        mask, nullptr, ksc, nullptr, 1, Nseq);

    // ---- gk = einsum(ksc, k) ; u = max_g(gk * v) (bf16) ----
    einsum_gdg<<<Bb * 16, TPB, 0, stream>>>(ksc, kf, gk, Nseq);
    maxcombine<<<(unsigned)(((size_t)M * DM) / TPB), TPB, 0, stream>>>(vf, gk, ub, Nseq);

    // ---- r = u @ Wt + bt + q ----
    gemm_bf16_wmma<<<gridBig, TPB, 0, stream>>>(ub, WtT, bt, M, K, DM, 1.0f,
        nullptr, qf, (float*)d_out, nullptr, 0, Nseq);
}
